// CrossAttentionBlock_28733331210564
// MI455X (gfx1250) — matrix operand order repaired and checked by reading, not compiled, not run on hardware
//
#include <hip/hip_runtime.h>
#include <hip/hip_bf16.h>

#define DIM 512
#define HEADS 8
#define DHEAD 64

typedef __attribute__((ext_vector_type(16))) __bf16 v16bf;
typedef __attribute__((ext_vector_type(8)))  float  v8f;
typedef __attribute__((ext_vector_type(4)))  int    v4i;

union Frag16 { v16bf v; uint4 u[2]; unsigned short s[16]; };

__device__ __forceinline__ unsigned short f2bf(float f) {
  unsigned int u = __float_as_uint(f);
  unsigned int r = u + 0x7FFFu + ((u >> 16) & 1u);   // round-to-nearest-even
  return (unsigned short)(r >> 16);
}

__device__ __forceinline__ v8f wmma_bf16(const Frag16& a, const Frag16& b, v8f c) {
  return __builtin_amdgcn_wmma_f32_16x16x32_bf16(false, a.v, false, b.v,
                                                 (short)0, c, false, false);
}

// ---- async global->LDS (CDNA5 ASYNCcnt path), sync-copy fallback ----------
#if __has_builtin(__builtin_amdgcn_global_load_async_to_lds_b128)
#define HAVE_ASYNC_LDS 1
#else
#define HAVE_ASYNC_LDS 0
#endif

typedef __attribute__((address_space(1))) v4i gv4i;  // global int4
typedef __attribute__((address_space(3))) v4i lv4i;  // LDS int4

__device__ __forceinline__ void async_b128(const unsigned short* g,
                                           unsigned short* l) {
#if HAVE_ASYNC_LDS
  __builtin_amdgcn_global_load_async_to_lds_b128((gv4i*)g, (lv4i*)l, 0, 0);
#else
  *(uint4*)l = *(const uint4*)g;
#endif
}

__device__ __forceinline__ void wait_async() {
#if HAVE_ASYNC_LDS
#if __has_builtin(__builtin_amdgcn_s_wait_asynccnt)
  __builtin_amdgcn_s_wait_asynccnt(0);
#else
  asm volatile("s_wait_asynccnt 0" ::: "memory");
#endif
#endif
}

// ---- 16-lane xor butterflies via v_permlane16_b32 (VALU, co-issues w/ WMMA)
__device__ __forceinline__ float pl16x(float v, unsigned sl, unsigned sh) {
#if __has_builtin(__builtin_amdgcn_permlane16)
  return __uint_as_float(__builtin_amdgcn_permlane16(
      __float_as_uint(v), __float_as_uint(v), sl, sh, false, false));
#else
  (void)sl; (void)sh;
  return v;
#endif
}

__device__ __forceinline__ float grp16_max(float v) {
#if __has_builtin(__builtin_amdgcn_permlane16)
  v = fmaxf(v, pl16x(v, 0x67452301u, 0xEFCDAB89u));  // xor 1
  v = fmaxf(v, pl16x(v, 0x54761032u, 0xDCFE98BAu));  // xor 2
  v = fmaxf(v, pl16x(v, 0x32107654u, 0xBA98FEDCu));  // xor 4
  v = fmaxf(v, pl16x(v, 0xFEDCBA98u, 0x76543210u));  // xor 8
#else
  #pragma unroll
  for (int m = 1; m < 16; m <<= 1) v = fmaxf(v, __shfl_xor(v, m, 32));
#endif
  return v;
}

__device__ __forceinline__ float grp16_sum(float v) {
#if __has_builtin(__builtin_amdgcn_permlane16)
  v += pl16x(v, 0x67452301u, 0xEFCDAB89u);
  v += pl16x(v, 0x54761032u, 0xDCFE98BAu);
  v += pl16x(v, 0x32107654u, 0xBA98FEDCu);
  v += pl16x(v, 0xFEDCBA98u, 0x76543210u);
#else
  #pragma unroll
  for (int m = 1; m < 16; m <<= 1) v += __shfl_xor(v, m, 32);
#endif
  return v;
}

// ---------------------------------------------------------------- fp32->bf16
__global__ __launch_bounds__(256) void cvt_bf16_kernel(
    const float* __restrict__ in, unsigned short* __restrict__ out, int n) {
  int i = blockIdx.x * blockDim.x + threadIdx.x;
  int stride = gridDim.x * blockDim.x;
  for (; i < n; i += stride) out[i] = f2bf(in[i]);
}

// ------------------------------------------------- fp32 KxN -> bf16 NxK (^T)
__global__ __launch_bounds__(256) void cvt_bf16_t_kernel(
    const float* __restrict__ in, unsigned short* __restrict__ out,
    int K, int N) {
  __shared__ float tile[32][33];
  const int tx = threadIdx.x & 31, ty = threadIdx.x >> 5;  // 32 x 8
  const int n0 = blockIdx.x * 32, k0 = blockIdx.y * 32;
  #pragma unroll
  for (int r = 0; r < 4; ++r)
    tile[ty + 8 * r][tx] = in[(size_t)(k0 + ty + 8 * r) * N + n0 + tx];
  __syncthreads();
  #pragma unroll
  for (int r = 0; r < 4; ++r)
    out[(size_t)(n0 + ty + 8 * r) * K + k0 + tx] = f2bf(tile[tx][ty + 8 * r]);
}

// ---------------------------------------------------------------- LayerNorm
__global__ __launch_bounds__(256) void ln_kernel(
    const float* __restrict__ x,
    const float* __restrict__ g1, const float* __restrict__ b1,
    const float* __restrict__ g2, const float* __restrict__ b2,
    int twice, unsigned short* __restrict__ out) {
  __shared__ float sh[16];
  const int row = blockIdx.x, t = threadIdx.x;
  const float* xr = x + (size_t)row * DIM;
  float v0 = xr[t], v1 = xr[t + 256];

  auto norm = [&](float& a, float& c, const float* g, const float* bb) {
    float s = a + c, q = a * a + c * c;
    s = grp16_sum(s); q = grp16_sum(q);
    s += __shfl_xor(s, 16, 32); q += __shfl_xor(q, 16, 32);
    const int wave = t >> 5, lane = t & 31;
    __syncthreads();
    if (lane == 0) { sh[wave] = s; sh[8 + wave] = q; }
    __syncthreads();
    s = 0.f; q = 0.f;
    #pragma unroll
    for (int i = 0; i < 8; ++i) { s += sh[i]; q += sh[8 + i]; }
    const float mean = s * (1.0f / DIM);
    const float var  = q * (1.0f / DIM) - mean * mean;
    const float rstd = rsqrtf(var + 1e-5f);
    a = (a - mean) * rstd * g[t]       + bb[t];
    c = (c - mean) * rstd * g[t + 256] + bb[t + 256];
  };
  norm(v0, v1, g1, b1);
  if (twice) norm(v0, v1, g2, b2);
  out[(size_t)row * DIM + t]       = f2bf(v0);
  out[(size_t)row * DIM + t + 256] = f2bf(v1);
}

// ---------------------------------------------------------------- GEMM
// C[M,N] = A[M,K](bf16,row-major) * Bt[N,K](bf16, pre-transposed weights).
// MODE 0: out = bf16(acc);  MODE 1: outf = resid + acc + bias (fp32);
// MODE 2: out = bf16(gelu(acc + bias)).
// 256 thr = 8 waves; block tile 128x128; wave tile 32x64 (8 WMMA / K-step);
// K-step 32; double-buffered LDS with async global->LDS staging.
template <int MODE>
__global__ __launch_bounds__(256) void gemm_bf16_kernel(
    const unsigned short* __restrict__ A,
    const unsigned short* __restrict__ Bt,
    const float* __restrict__ bias,
    const float* __restrict__ resid,
    unsigned short* __restrict__ outb,
    float* __restrict__ outf,
    int M, int N, int K) {
  __shared__ __align__(16) unsigned short As[2][128 * 32];   // [m][k]
  __shared__ __align__(16) unsigned short Bs[2][128 * 32];   // [n][k]
  const int t = threadIdx.x;
  const int lane = t & 31, wave = t >> 5;
  const int wm = wave & 3, wn = wave >> 2;
  const int m = lane & 15, half = lane >> 4;
  const int bm = blockIdx.y * 128;
  const int bn = blockIdx.x * 128;

  v8f acc[2][4];
  #pragma unroll
  for (int s = 0; s < 2; ++s)
    #pragma unroll
    for (int i = 0; i < 4; ++i)
      #pragma unroll
      for (int e = 0; e < 8; ++e) acc[s][i][e] = 0.0f;

  const int srow = t >> 1, scol = (t & 1) * 16;  // 128x32 tile, 32B/thread

  const unsigned short* aSrc = &A[(size_t)(bm + srow) * K + scol];
  const unsigned short* bSrc = &Bt[(size_t)(bn + srow) * K + scol];

  auto stage = [&](int buf, int kk) {
    async_b128(aSrc + kk,     &As[buf][srow * 32 + scol]);
    async_b128(aSrc + kk + 8, &As[buf][srow * 32 + scol + 8]);
    async_b128(bSrc + kk,     &Bs[buf][srow * 32 + scol]);
    async_b128(bSrc + kk + 8, &Bs[buf][srow * 32 + scol + 8]);
    if (kk + 32 < K) {  // prime L2/WGP$ for the tile after next
      __builtin_prefetch((const void*)(aSrc + kk + 32), 0, 1);
      __builtin_prefetch((const void*)(bSrc + kk + 32), 0, 1);
    }
  };

  stage(0, 0);
  int cur = 0;
  for (int k0 = 0; k0 < K; k0 += 32) {
    wait_async();
    __syncthreads();
    if (k0 + 32 < K) stage(cur ^ 1, k0 + 32);

    Frag16 fa[2];  // A frags: rows wm*32 + sm*16 + m
    #pragma unroll
    for (int sm = 0; sm < 2; ++sm) {
      const int ar = (wm * 32 + sm * 16 + m) * 32;
      fa[sm].u[0] = *(const uint4*)&As[cur][ar + half * 8];
      fa[sm].u[1] = *(const uint4*)&As[cur][ar + 16 + half * 8];
    }
    #pragma unroll
    for (int nt = 0; nt < 4; ++nt) {
      Frag16 fb;  // B frag: col n = lane%16, K = half*8 + e (e<8), 16 + half*8 + (e-8)
      const int ncol = wn * 64 + nt * 16 + m;
      fb.u[0] = *(const uint4*)&Bs[cur][ncol * 32 + half * 8];
      fb.u[1] = *(const uint4*)&Bs[cur][ncol * 32 + 16 + half * 8];
      acc[0][nt] = wmma_bf16(fa[0], fb, acc[0][nt]);
      acc[1][nt] = wmma_bf16(fa[1], fb, acc[1][nt]);
    }
    cur ^= 1;
  }

  #pragma unroll
  for (int sm = 0; sm < 2; ++sm) {
    #pragma unroll
    for (int nt = 0; nt < 4; ++nt) {
      const int col = bn + wn * 64 + nt * 16 + m;
      const float bv = (MODE == 0) ? 0.0f : bias[col];
      #pragma unroll
      for (int r = 0; r < 8; ++r) {
        const int row = bm + wm * 32 + sm * 16 + r + 8 * half;
        float v = acc[sm][nt][r] + bv;
        const size_t idx = (size_t)row * N + col;
        if (MODE == 1) {
          outf[idx] = resid[idx] + v;
        } else if (MODE == 2) {
          v = 0.5f * v * (1.0f + erff(v * 0.70710678118654752f));
          outb[idx] = f2bf(v);
        } else {
          outb[idx] = f2bf(v);
        }
      }
    }
  }
}

// ---------------------------------------------------------------- Attention
// Flash-style; 128 thr = 4 waves, each wave a 16-row Q tile of one (b,h).
// Double-buffered K/V LDS staging (K via async), one barrier per 32-key block.
__global__ __launch_bounds__(128) void attn_kernel(
    const unsigned short* __restrict__ qp, int qs,
    const unsigned short* __restrict__ kp, int ks,
    const unsigned short* __restrict__ vp, int vs,
    unsigned short* __restrict__ op, int os,
    int nq, int nk) {
  __shared__ __align__(16) unsigned short Kb[2][32 * 64];   // [j][d]
  __shared__ __align__(16) unsigned short Vt[2][64 * 32];   // [d][j]
  __shared__ __align__(16) unsigned short Pl[4 * 16 * 32];

  const int t = threadIdx.x;
  const int lane = t & 31, wave = t >> 5;
  const int m = lane & 15, half = lane >> 4;
  const int b = blockIdx.z, h = blockIdx.y;
  const int q0 = blockIdx.x * 64 + wave * 16;
  const int hoff = h * DHEAD;
  const float scale = 0.125f;  // DHEAD^-0.5

  Frag16 fq[2];  // Q rows, d-slices [0,32) and [32,64)
  {
    const unsigned short* qr = qp + (size_t)(b * nq + q0 + m) * qs + hoff;
    #pragma unroll
    for (int d = 0; d < 2; ++d) {
      fq[d].u[0] = *(const uint4*)(qr + d * 32 + half * 8);
      fq[d].u[1] = *(const uint4*)(qr + d * 32 + 16 + half * 8);
    }
  }

  float mrun[8], lrun[8];
  v8f o[4];
  #pragma unroll
  for (int r = 0; r < 8; ++r) { mrun[r] = -1e30f; lrun[r] = 0.0f; }
  #pragma unroll
  for (int i = 0; i < 4; ++i)
    #pragma unroll
    for (int e = 0; e < 8; ++e) o[i][e] = 0.0f;

  const int lrow = t >> 2;        // 0..31
  const int lcol = (t & 3) * 16;  // 0,16,32,48

  auto stageKV = [&](int buf, int jb) {
    const unsigned short* ksrc =
        kp + (size_t)(b * nk + jb + lrow) * ks + hoff + lcol;
    async_b128(ksrc,     &Kb[buf][lrow * 64 + lcol]);
    async_b128(ksrc + 8, &Kb[buf][lrow * 64 + lcol + 8]);
    const unsigned short* vsrc =
        vp + (size_t)(b * nk + jb + lrow) * vs + hoff + lcol;
    unsigned short tmp[16];
    *(uint4*)&tmp[0] = *(const uint4*)(vsrc);
    *(uint4*)&tmp[8] = *(const uint4*)(vsrc + 8);
    #pragma unroll
    for (int i = 0; i < 16; ++i) Vt[buf][(lcol + i) * 32 + lrow] = tmp[i];
    if (jb + 32 < nk) {
      __builtin_prefetch((const void*)(ksrc + 32 * ks), 0, 1);
      __builtin_prefetch((const void*)(vsrc + 32 * vs), 0, 1);
    }
  };

  stageKV(0, 0);
  int cur = 0;
  for (int jb = 0; jb < nk; jb += 32) {
    wait_async();
    __syncthreads();
    if (jb + 32 < nk) stageKV(cur ^ 1, jb + 32);

    // S = scale * Q K^T  -> two 16x16 tiles (keys jb..jb+31)
    v8f s[2];
    #pragma unroll
    for (int c = 0; c < 2; ++c) {
      v8f sc;
      #pragma unroll
      for (int e = 0; e < 8; ++e) sc[e] = 0.0f;
      #pragma unroll
      for (int kslice = 0; kslice < 2; ++kslice) {
        Frag16 fk;  // B = K^T: col n = key (c*16+m), K = kslice*32 + table order
        const int krow = c * 16 + m;
        fk.u[0] =
            *(const uint4*)&Kb[cur][krow * 64 + kslice * 32 + half * 8];
        fk.u[1] =
            *(const uint4*)&Kb[cur][krow * 64 + kslice * 32 + 16 + half * 8];
        sc = wmma_bf16(fq[kslice], fk, sc);
      }
      #pragma unroll
      for (int e = 0; e < 8; ++e) sc[e] *= scale;
      s[c] = sc;
    }

    // online softmax over this 32-key block (row = r + 8*half)
    unsigned short* pw = &Pl[wave * 16 * 32];
    #pragma unroll
    for (int r = 0; r < 8; ++r) {
      const float mb  = grp16_max(fmaxf(s[0][r], s[1][r]));
      const float mn  = fmaxf(mrun[r], mb);
      const float fac = __expf(mrun[r] - mn);
      const float p0  = __expf(s[0][r] - mn);
      const float p1  = __expf(s[1][r] - mn);
      const float ls  = grp16_sum(p0 + p1);
      lrun[r] = lrun[r] * fac + ls;
      mrun[r] = mn;
      #pragma unroll
      for (int i = 0; i < 4; ++i) o[i][r] *= fac;
      const int prow = r + 8 * half;  // C layout -> LDS row-major
      pw[prow * 32 + m]      = f2bf(p0);
      pw[prow * 32 + 16 + m] = f2bf(p1);
    }
    asm volatile("s_wait_dscnt 0" ::: "memory");  // P visible to own wave

    // O += P(16x32) * V(32x64)
    Frag16 fp;
    fp.u[0] = *(const uint4*)&pw[m * 32 + half * 8];
    fp.u[1] = *(const uint4*)&pw[m * 32 + 16 + half * 8];
    #pragma unroll
    for (int dt = 0; dt < 4; ++dt) {
      Frag16 fv;  // B = V: col n = d (dt*16+m), K = j in table order
      const int dcol = dt * 16 + m;
      fv.u[0] = *(const uint4*)&Vt[cur][dcol * 32 + half * 8];
      fv.u[1] = *(const uint4*)&Vt[cur][dcol * 32 + 16 + half * 8];
      o[dt] = wmma_bf16(fp, fv, o[dt]);
    }
    cur ^= 1;
  }

  #pragma unroll
  for (int dt = 0; dt < 4; ++dt) {
    #pragma unroll
    for (int r = 0; r < 8; ++r) {
      const int row = q0 + r + 8 * half;
      const int col = hoff + dt * 16 + m;
      op[(size_t)(b * nq + row) * os + col] = f2bf(o[dt][r] / lrun[r]);
    }
  }
}

// ---------------------------------------------------------------- driver
extern "C" void kernel_launch(void* const* d_in, const int* in_sizes, int n_in,
                              void* d_out, int out_size, void* d_ws,
                              size_t ws_size, hipStream_t stream) {
  (void)in_sizes; (void)n_in; (void)out_size; (void)ws_size;
  const float* x       = (const float*)d_in[0];
  const float* y       = (const float*)d_in[1];
  const float* sa_g    = (const float*)d_in[2];
  const float* sa_b    = (const float*)d_in[3];
  const float* sa_wqkv = (const float*)d_in[4];
  const float* sa_wo   = (const float*)d_in[5];
  const float* sa_bo   = (const float*)d_in[6];
  const float* n1_g    = (const float*)d_in[7];
  const float* n1_b    = (const float*)d_in[8];
  const float* ca_g    = (const float*)d_in[9];
  const float* ca_b    = (const float*)d_in[10];
  const float* ca_wq   = (const float*)d_in[11];
  const float* ca_wkv  = (const float*)d_in[12];
  const float* ca_wo   = (const float*)d_in[13];
  const float* ca_bo   = (const float*)d_in[14];
  const float* ff_g    = (const float*)d_in[15];
  const float* ff_b    = (const float*)d_in[16];
  const float* ff_w1   = (const float*)d_in[17];
  const float* ff_b1   = (const float*)d_in[18];
  const float* ff_w2   = (const float*)d_in[19];
  const float* ff_b2   = (const float*)d_in[20];

  const int B = 4, NX = 2048, NY = 1024;
  const int MX = B * NX;  // 8192 query rows
  const int MY = B * NY;  // 4096 kv rows

  char* p = (char*)d_ws;
  auto carve = [&](size_t bytes) -> void* {
    void* r = (void*)p;
    p += (bytes + 255) & ~(size_t)255;
    return r;
  };
  // transposed bf16 weights ([N][K])
  unsigned short* wqkv_t = (unsigned short*)carve((size_t)1536 * DIM * 2);
  unsigned short* wo_t   = (unsigned short*)carve((size_t)512 * DIM * 2);
  unsigned short* wq_t   = (unsigned short*)carve((size_t)512 * DIM * 2);
  unsigned short* wkv_t  = (unsigned short*)carve((size_t)1024 * DIM * 2);
  unsigned short* cawo_t = (unsigned short*)carve((size_t)512 * DIM * 2);
  unsigned short* w1_t   = (unsigned short*)carve((size_t)2048 * DIM * 2);
  unsigned short* w2_t   = (unsigned short*)carve((size_t)DIM * 2048 * 2);
  // activations
  unsigned short* y_b   = (unsigned short*)carve((size_t)MY * DIM * 2);
  unsigned short* xn_b  = (unsigned short*)carve((size_t)MX * DIM * 2);
  unsigned short* qkv_b = (unsigned short*)carve((size_t)MX * 1536 * 2);
  unsigned short* ao_b  = (unsigned short*)carve((size_t)MX * DIM * 2);
  unsigned short* q2_b  = (unsigned short*)carve((size_t)MX * DIM * 2);
  unsigned short* kv_b  = (unsigned short*)carve((size_t)MY * 1024 * 2);
  unsigned short* h_b   = (unsigned short*)carve((size_t)MX * 2048 * 2);
  float*          xbuf  = (float*)carve((size_t)MX * DIM * 4);
  float*          xout  = (float*)d_out;

  // one-shot weight transpose+convert; y plain convert
  cvt_bf16_t_kernel<<<dim3(1536 / 32, DIM / 32), 256, 0, stream>>>(sa_wqkv, wqkv_t, DIM, 1536);
  cvt_bf16_t_kernel<<<dim3(512 / 32, DIM / 32), 256, 0, stream>>>(sa_wo, wo_t, DIM, 512);
  cvt_bf16_t_kernel<<<dim3(512 / 32, DIM / 32), 256, 0, stream>>>(ca_wq, wq_t, DIM, 512);
  cvt_bf16_t_kernel<<<dim3(1024 / 32, DIM / 32), 256, 0, stream>>>(ca_wkv, wkv_t, DIM, 1024);
  cvt_bf16_t_kernel<<<dim3(512 / 32, DIM / 32), 256, 0, stream>>>(ca_wo, cawo_t, DIM, 512);
  cvt_bf16_t_kernel<<<dim3(2048 / 32, DIM / 32), 256, 0, stream>>>(ff_w1, w1_t, DIM, 2048);
  cvt_bf16_t_kernel<<<dim3(512 / 32, 2048 / 32), 256, 0, stream>>>(ff_w2, w2_t, 2048, 512);
  cvt_bf16_kernel<<<2048, 256, 0, stream>>>(y, y_b, MY * DIM);

  // ---- self attention ----
  ln_kernel<<<MX, 256, 0, stream>>>(x, sa_g, sa_b, nullptr, nullptr, 0, xn_b);
  gemm_bf16_kernel<0><<<dim3(1536 / 128, MX / 128), 256, 0, stream>>>(
      xn_b, wqkv_t, nullptr, nullptr, qkv_b, nullptr, MX, 1536, DIM);
  attn_kernel<<<dim3(NX / 64, HEADS, B), 128, 0, stream>>>(
      qkv_b, 1536, qkv_b + 512, 1536, qkv_b + 1024, 1536, ao_b, 512, NX, NX);
  gemm_bf16_kernel<1><<<dim3(512 / 128, MX / 128), 256, 0, stream>>>(
      ao_b, wo_t, sa_bo, x, nullptr, xbuf, MX, 512, DIM);

  // ---- cross attention ----
  ln_kernel<<<MX, 256, 0, stream>>>(xbuf, n1_g, n1_b, ca_g, ca_b, 1, xn_b);
  gemm_bf16_kernel<0><<<dim3(512 / 128, MX / 128), 256, 0, stream>>>(
      xn_b, wq_t, nullptr, nullptr, q2_b, nullptr, MX, 512, DIM);
  gemm_bf16_kernel<0><<<dim3(1024 / 128, MY / 128), 256, 0, stream>>>(
      y_b, wkv_t, nullptr, nullptr, kv_b, nullptr, MY, 1024, DIM);
  attn_kernel<<<dim3(NX / 64, HEADS, B), 128, 0, stream>>>(
      q2_b, 512, kv_b, 1024, kv_b + 512, 1024, ao_b, 512, NX, NY);
  gemm_bf16_kernel<1><<<dim3(512 / 128, MX / 128), 256, 0, stream>>>(
      ao_b, cawo_t, ca_bo, xbuf, nullptr, xbuf, MX, 512, DIM);

  // ---- MLP ----
  ln_kernel<<<MX, 256, 0, stream>>>(xbuf, ff_g, ff_b, nullptr, nullptr, 0, xn_b);
  gemm_bf16_kernel<2><<<dim3(2048 / 128, MX / 128), 256, 0, stream>>>(
      xn_b, w1_t, ff_b1, nullptr, h_b, nullptr, MX, 2048, DIM);
  gemm_bf16_kernel<1><<<dim3(512 / 128, MX / 128), 256, 0, stream>>>(
      h_b, w2_t, ff_b2, xbuf, nullptr, xout, MX, 512, 2048);
}
